// PointNet2PartSegmentNet_69698729279693
// MI455X (gfx1250) — matrix operand order repaired and checked by reading, not compiled, not run on hardware
//
#include <hip/hip_runtime.h>
#include <hip/hip_bf16.h>

// PointNet++ part-seg for MI455X (gfx1250, wave32).
// All dense layers run through v_wmma_f32_16x16x32_f16 (f16 in, f32 acc).
// Activations are stored with K padded to 32 and weights pre-packed into the
// WMMA B-fragment layout (N padded to 32), so the GEMM inner loop is pure
// b128 loads + WMMA with EXEC all-ones. Each wave computes a 32x32 output
// tile: 4 WMMAs per 8 b128 loads.
// Workspace requirement: ~110 MB (two f16 ping-pong GEMM buffers dominate).

typedef __attribute__((ext_vector_type(16))) _Float16 v16h;
typedef __attribute__((ext_vector_type(8)))  _Float16 v8h;
typedef __attribute__((ext_vector_type(8)))  float    v8f;

#define DEV_INF __builtin_inff()

static inline int pad32(int x) { return (x + 31) & ~31; }

// ---------------------------------------------------------------- init / transpose
__global__ void init_pos_kernel(const float* __restrict__ data, // (3,N) slice
                                float* __restrict__ P0, float* __restrict__ X0, int N) {
  int t = blockIdx.x * blockDim.x + threadIdx.x;
  if (t >= N * 3) return;
  int n = t / 3, c = t % 3;
  float v = data[(size_t)c * N + n];
  P0[t] = v;
  X0[t] = v;
}

// ---------------------------------------------------------------- farthest point sampling
__global__ void fps_kernel(const float* __restrict__ pos, int n, int m,
                           int* __restrict__ idx) {
  __shared__ float mind[4096];
  __shared__ float rv[256];
  __shared__ int   ri[256];
  __shared__ int   s_cur;
  const int tid = threadIdx.x, T = blockDim.x;
  for (int j = tid; j < n; j += T) mind[j] = DEV_INF;
  if (tid == 0) { s_cur = 0; idx[0] = 0; }
  __syncthreads();
  for (int t = 1; t < m; ++t) {
    int cur = s_cur;
    float cx = pos[cur * 3 + 0], cy = pos[cur * 3 + 1], cz = pos[cur * 3 + 2];
    float bv = -1.0f; int bj = n - 1;
    for (int j = tid; j < n; j += T) {
      float dx = pos[j * 3 + 0] - cx;
      float dy = pos[j * 3 + 1] - cy;
      float dz = pos[j * 3 + 2] - cz;
      float d = dx * dx + dy * dy + dz * dz;
      float mn = fminf(mind[j], d);
      mind[j] = mn;
      if (mn > bv) { bv = mn; bj = j; }     // strict > keeps lowest index on tie
    }
    rv[tid] = bv; ri[tid] = bj;
    __syncthreads();
    for (int s = T >> 1; s > 0; s >>= 1) {
      if (tid < s) {
        if (rv[tid + s] > rv[tid] ||
            (rv[tid + s] == rv[tid] && ri[tid + s] < ri[tid])) {
          rv[tid] = rv[tid + s]; ri[tid] = ri[tid + s];
        }
      }
      __syncthreads();
    }
    if (tid == 0) { s_cur = ri[0]; idx[t] = ri[0]; }
    __syncthreads();
  }
}

__global__ void gather_pos_kernel(const float* __restrict__ pos,
                                  const int* __restrict__ idx,
                                  float* __restrict__ q, int m) {
  int t = blockIdx.x * blockDim.x + threadIdx.x;
  if (t >= m * 3) return;
  int i = t / 3, c = t % 3;
  q[t] = pos[(size_t)idx[i] * 3 + c];
}

// ---------------------------------------------------------------- k-NN (sorted insertion, stable ties)
template <int KK>
__global__ void knn_kernel(const float* __restrict__ qp, int nq,
                           const float* __restrict__ sp, int ns,
                           int* __restrict__ nbr, float* __restrict__ d2o) {
  int i = blockIdx.x * blockDim.x + threadIdx.x;
  if (i >= nq) return;
  float qx = qp[i * 3 + 0], qy = qp[i * 3 + 1], qz = qp[i * 3 + 2];
  float bd[KK]; int bi[KK];
  for (int j = 0; j < KK; ++j) { bd[j] = DEV_INF; bi[j] = 0; }
  for (int n = 0; n < ns; ++n) {
    float dx = sp[n * 3 + 0] - qx;
    float dy = sp[n * 3 + 1] - qy;
    float dz = sp[n * 3 + 2] - qz;
    float d = dx * dx + dy * dy + dz * dz;
    if (d < bd[KK - 1]) {
      int p = KK - 1;
      while (p > 0 && bd[p - 1] > d) { bd[p] = bd[p - 1]; bi[p] = bi[p - 1]; --p; }
      bd[p] = d; bi[p] = n;
    }
  }
  for (int j = 0; j < KK; ++j) {
    nbr[(size_t)i * KK + j] = bi[j];
    d2o[(size_t)i * KK + j] = bd[j];
  }
}

// ---------------------------------------------------------------- weight packing into WMMA B-fragment layout
// Wp[((kt*ntN + nt)*32 + lane)*16 + i] = W[kt*32 + (i>>3)*16 + (lane>>4)*8 + (i&7)][nt*16 + (lane&15)]
// N padded to 32 so the GEMM can consume two adjacent 16-wide N tiles per wave.
__global__ void pack_w_kernel(const float* __restrict__ W, _Float16* __restrict__ Wp,
                              int K, int N) {
  const int Kp = (K + 31) & ~31, Np = (N + 31) & ~31;
  const int ntN = Np >> 4;
  size_t tot = (size_t)(Kp >> 5) * ntN * 32 * 16;
  size_t t = (size_t)blockIdx.x * blockDim.x + threadIdx.x;
  if (t >= tot) return;
  int i = (int)(t & 15);
  size_t r = t >> 4;
  int lane = (int)(r & 31);
  r >>= 5;
  int nt = (int)(r % ntN);
  int kt = (int)(r / ntN);
  int k = kt * 32 + (i >> 3) * 16 + (lane >> 4) * 8 + (i & 7);
  int n = nt * 16 + (lane & 15);
  float v = (k < K && n < N) ? W[(size_t)k * N + n] : 0.0f;
  Wp[t] = (_Float16)v;
}

// ---------------------------------------------------------------- SA row build (padded lda, zero pad cols)
__global__ void sa_build_kernel(const float* __restrict__ x, int Cx,
                                const float* __restrict__ pos,
                                const float* __restrict__ q,
                                const int* __restrict__ nbr,
                                int m, int lda, _Float16* __restrict__ rows) {
  const int K = 64;
  const int Ct = Cx + 3;
  size_t t = (size_t)blockIdx.x * blockDim.x + threadIdx.x;
  size_t tot = (size_t)m * K * lda;
  if (t >= tot) return;
  int c = (int)(t % lda);
  size_t r = t / lda;
  float v = 0.0f;
  if (c < Ct) {
    int i = (int)(r / K);
    int nb = nbr[r];
    if (c < Cx) v = x[(size_t)nb * Cx + c];
    else        v = pos[(size_t)nb * 3 + (c - Cx)] - q[(size_t)i * 3 + (c - Cx)];
  }
  rows[t] = (_Float16)v;
}

// ---------------------------------------------------------------- WMMA GEMM: C = act(A @ W + b)
// One wave computes a 32x32 tile: 4 WMMAs per k-step sharing 2 A-fragments
// and 2 B-fragments (8 b128 loads). A: f16, row stride lda (multiple of 32;
// pad garbage is cancelled by zero-padded packed weights).
template <typename TC>
__global__ __launch_bounds__(32)
void gemm_wmma_kernel(const _Float16* __restrict__ A, int lda,
                      const _Float16* __restrict__ Wp,
                      const float* __restrict__ bias,
                      TC* __restrict__ C, int ldc,
                      int M, int Kp, int N, int relu) {
  const int tm = blockIdx.x * 32;
  const int tn = blockIdx.y * 32;
  const int lane = threadIdx.x & 31;
  const int half = lane >> 4, lan = lane & 15;

  const _Float16* a0p = A + (size_t)(tm + lan) * lda + half * 8;
  const _Float16* a1p = a0p + (size_t)16 * lda;
  const _Float16* bp  = Wp + ((size_t)blockIdx.y * 2 * 32 + lane) * 16;
  const size_t bstride = (size_t)gridDim.y * 2 * 32 * 16;   // per 32-K tile

  v8f acc00 = {}, acc01 = {}, acc10 = {}, acc11 = {};
  for (int k0 = 0; k0 < Kp; k0 += 32) {
    v8h al0 = *(const v8h*)(a0p + k0);
    v8h ah0 = *(const v8h*)(a0p + k0 + 16);
    v8h al1 = *(const v8h*)(a1p + k0);
    v8h ah1 = *(const v8h*)(a1p + k0 + 16);
    v8h b0l = *(const v8h*)(bp);
    v8h b0h = *(const v8h*)(bp + 8);
    v8h b1l = *(const v8h*)(bp + 512);
    v8h b1h = *(const v8h*)(bp + 520);
    bp += bstride;
    v16h a0, a1, b0, b1;
#pragma unroll
    for (int i = 0; i < 8; ++i) {
      a0[i] = al0[i]; a0[8 + i] = ah0[i];
      a1[i] = al1[i]; a1[8 + i] = ah1[i];
      b0[i] = b0l[i]; b0[8 + i] = b0h[i];
      b1[i] = b1l[i]; b1[8 + i] = b1h[i];
    }
    acc00 = __builtin_amdgcn_wmma_f32_16x16x32_f16(false, a0, false, b0,
                                                   (short)0, acc00, false, false);
    acc01 = __builtin_amdgcn_wmma_f32_16x16x32_f16(false, a0, false, b1,
                                                   (short)0, acc01, false, false);
    acc10 = __builtin_amdgcn_wmma_f32_16x16x32_f16(false, a1, false, b0,
                                                   (short)0, acc10, false, false);
    acc11 = __builtin_amdgcn_wmma_f32_16x16x32_f16(false, a1, false, b1,
                                                   (short)0, acc11, false, false);
  }

  const int col0 = tn + lan;
  const int col1 = tn + 16 + lan;
  const float bv0 = (bias && col0 < N) ? bias[col0] : 0.0f;
  const float bv1 = (bias && col1 < N) ? bias[col1] : 0.0f;
#pragma unroll
  for (int v = 0; v < 8; ++v) {
    int row0 = tm + half * 8 + v;        // D: lanes 0-15 M=v, lanes 16-31 M=8+v
    int row1 = row0 + 16;
    float r00 = acc00[v] + bv0, r01 = acc01[v] + bv1;
    float r10 = acc10[v] + bv0, r11 = acc11[v] + bv1;
    if (relu) {
      r00 = fmaxf(r00, 0.0f); r01 = fmaxf(r01, 0.0f);
      r10 = fmaxf(r10, 0.0f); r11 = fmaxf(r11, 0.0f);
    }
    if (row0 < M) {
      if (col0 < N) C[(size_t)row0 * ldc + col0] = (TC)r00;
      if (col1 < N) C[(size_t)row0 * ldc + col1] = (TC)r01;
    }
    if (row1 < M) {
      if (col0 < N) C[(size_t)row1 * ldc + col0] = (TC)r10;
      if (col1 < N) C[(size_t)row1 * ldc + col1] = (TC)r11;
    }
  }
}

// ---------------------------------------------------------------- masked max-pool over K neighbors
__global__ void maxpool_kernel(const _Float16* __restrict__ H,
                               const float* __restrict__ d2, float r2,
                               float* __restrict__ X, int m, int C, int K) {
  size_t t = (size_t)blockIdx.x * blockDim.x + threadIdx.x;
  if (t >= (size_t)m * C) return;
  int i = (int)(t / C), c = (int)(t % C);
  float best = -DEV_INF;
  for (int j = 0; j < K; ++j) {
    if (d2[(size_t)i * K + j] <= r2) {
      float v = (float)H[((size_t)i * K + j) * C + c];
      best = fmaxf(best, v);
    }
  }
  X[t] = best;
}

// ---------------------------------------------------------------- FP row build (padded lda)
__global__ void fp_build_kernel(const float* __restrict__ inx, int Cin,
                                const float* __restrict__ skipx, int Cs,
                                const float* __restrict__ skipp,
                                const int* __restrict__ nbr,
                                const float* __restrict__ d2,
                                int k, int ns, int lda, _Float16* __restrict__ rows) {
  const int Ct = Cin + Cs + 3;
  size_t t = (size_t)blockIdx.x * blockDim.x + threadIdx.x;
  if (t >= (size_t)ns * lda) return;
  int i = (int)(t / lda), c = (int)(t % lda);
  float v = 0.0f;
  if (c < Cin) {
    float w[3]; float sum = 0.0f;
    for (int j = 0; j < k; ++j) {
      float d = sqrtf(fmaxf(d2[(size_t)i * k + j], 0.0f));
      d = fmaxf(d, 1e-10f);
      w[j] = 1.0f / d;
      sum += w[j];
    }
    float inv = 1.0f / (sum + 1e-16f);
    v = 0.0f;
    for (int j = 0; j < k; ++j)
      v += w[j] * inv * inx[(size_t)nbr[(size_t)i * k + j] * Cin + c];
  } else if (c < Cin + Cs) {
    v = skipx[(size_t)i * Cs + (c - Cin)];
  } else if (c < Ct) {
    v = skipp[(size_t)i * 3 + (c - Cin - Cs)];
  }
  rows[t] = (_Float16)v;
}

// ---------------------------------------------------------------- f32 -> f16 copy
__global__ void cvt_f16_kernel(const float* __restrict__ src,
                               _Float16* __restrict__ dst, int n) {
  int t = blockIdx.x * blockDim.x + threadIdx.x;
  if (t < n) dst[t] = (_Float16)src[t];
}

// ---------------------------------------------------------------- mean over NUM_CLASSES=50
__global__ void mean50_kernel(const float* __restrict__ logits,
                              float* __restrict__ out, int n) {
  int i = blockIdx.x * blockDim.x + threadIdx.x;
  if (i >= n) return;
  float s = 0.0f;
  for (int c = 0; c < 50; ++c) s += logits[(size_t)i * 50 + c];
  out[i] = s * (1.0f / 50.0f);
}

// ---------------------------------------------------------------- host-side GEMM dispatch
template <typename TC>
static inline void launch_gemm(const _Float16* A, int lda, const _Float16* Wp,
                               const float* bias, void* C, int ldc,
                               int M, int K, int N, int relu, hipStream_t s) {
  int Kp = pad32(K);
  dim3 g((unsigned)((M + 31) / 32), (unsigned)((N + 31) / 32));
  gemm_wmma_kernel<TC><<<g, dim3(32), 0, s>>>(A, lda, Wp, bias, (TC*)C, ldc,
                                              M, Kp, N, relu);
}

static inline size_t align256(size_t b) { return (b + 255) & ~(size_t)255; }

extern "C" void kernel_launch(void* const* d_in, const int* in_sizes, int n_in,
                              void* d_out, int out_size, void* d_ws, size_t ws_size,
                              hipStream_t stream) {
  (void)in_sizes; (void)n_in; (void)out_size; (void)ws_size;
  const float* data = (const float*)d_in[0];
  float* out = (float*)d_out;

  // ---- static shapes
  const int B = 4, NPTS = 4096;
  const int MS[5]  = {4096, 3277, 2294, 1377, 689};
  const int XCH[5] = {3, 64, 128, 256, 512};
  const float R2[4] = {0.025f * 0.025f, 0.05f * 0.05f, 0.1f * 0.1f, 0.2f * 0.2f};
  const int SA_IN[4] = {6, 67, 131, 259};
  const int SA_C[4][4] = {{16, 32, 32, 64}, {64, 64, 64, 128},
                          {128, 128, 128, 256}, {256, 256, 256, 512}};
  const int FP_BASE[4] = {33, 41, 49, 53};   // fp4, fp3, fp2, fp1 weight indices
  const int FP_NL[4]   = {4, 4, 2, 4};
  const int FP_IN[4]   = {771, 387, 323, 262};
  const int FP_CHS[4][4] = {{512, 256, 256, 256}, {256, 256, 256, 256},
                            {256, 256, 0, 0}, {256, 256, 256, 256}};
  const int FP_K[4] = {1, 1, 3, 3};
  const int CLS_D[5] = {256, 128, 64, 32, 50};

  // ---- workspace carve (per-batch reuse)
  char* ws = (char*)d_ws;
  size_t off = 0;
  auto alloc = [&](size_t bytes) -> char* {
    char* p = ws + off;
    off += align256(bytes);
    return p;
  };
  const size_t PING_ELEMS = 23000064;                 // max padded rows*lda
  _Float16* ping = (_Float16*)alloc(PING_ELEMS * 2);
  _Float16* pong = (_Float16*)alloc(PING_ELEMS * 2);
  int*   nbr = (int*)alloc((size_t)209728 * 4);       // 3277*64 max
  float* d2  = (float*)alloc((size_t)209728 * 4);
  int*   idx = (int*)alloc((size_t)4096 * 4);
  float* P[5]; float* X[5];
  for (int l = 0; l < 5; ++l) {
    P[l] = (float*)alloc((size_t)MS[l] * 3 * 4);
    X[l] = (float*)alloc((size_t)MS[l] * XCH[l] * 4);
  }
  float*    FA    = (float*)alloc((size_t)4096 * 256 * 4);
  float*    FB    = (float*)alloc((size_t)4096 * 256 * 4);
  float*    LOG   = (float*)alloc((size_t)4096 * 50 * 4);
  _Float16* FEATH = (_Float16*)alloc((size_t)4096 * 256 * 2);
  _Float16* CA    = (_Float16*)alloc((size_t)4096 * 128 * 2);
  _Float16* CB    = (_Float16*)alloc((size_t)4096 * 64 * 2);
  _Float16* WPK   = (_Float16*)alloc((size_t)4 * 1024 * 1024 * 2); // packed weights

  // ---- pack all weights once (layout matches gemm_wmma B-fragment reads)
  _Float16* wpSA[4][4]; _Float16* wpFP[4][4]; _Float16* wpCLS[2][4];
  {
    size_t woff = 0;
    auto pack = [&](int di, int K, int N) -> _Float16* {
      int Kp = pad32(K), Np = pad32(N);
      size_t tot = (size_t)(Kp >> 5) * (Np >> 4) * 32 * 16;
      _Float16* p = WPK + woff;
      woff += tot;
      pack_w_kernel<<<(unsigned)((tot + 255) / 256), 256, 0, stream>>>(
          (const float*)d_in[di], p, K, N);
      return p;
    };
    for (int l = 0; l < 4; ++l)
      for (int j = 0; j < 4; ++j)
        wpSA[l][j] = pack(1 + l * 8 + 2 * j,
                          j == 0 ? SA_IN[l] : SA_C[l][j - 1], SA_C[l][j]);
    for (int s = 0; s < 4; ++s)
      for (int j = 0; j < FP_NL[s]; ++j)
        wpFP[s][j] = pack(FP_BASE[s] + 2 * j,
                          j == 0 ? FP_IN[s] : FP_CHS[s][j - 1], FP_CHS[s][j]);
    for (int c = 0; c < 2; ++c)
      for (int j = 0; j < 4; ++j)
        wpCLS[c][j] = pack(61 + c * 5 + j, CLS_D[j], CLS_D[j + 1]);
  }

  for (int b = 0; b < B; ++b) {
    const float* db = data + (size_t)b * 3 * NPTS;
    init_pos_kernel<<<(NPTS * 3 + 255) / 256, 256, 0, stream>>>(db, P[0], X[0], NPTS);

    // ---------------- SA levels
    for (int l = 0; l < 4; ++l) {
      const int nsrc = MS[l], m = MS[l + 1];
      fps_kernel<<<1, 256, 0, stream>>>(P[l], nsrc, m, idx);
      gather_pos_kernel<<<(m * 3 + 255) / 256, 256, 0, stream>>>(P[l], idx, P[l + 1], m);
      knn_kernel<64><<<(m + 127) / 128, 128, 0, stream>>>(P[l + 1], m, P[l], nsrc, nbr, d2);

      const int Cx = XCH[l], Ct = Cx + 3;
      const int lda0 = pad32(Ct);
      const size_t rows = (size_t)m * 64;
      const size_t tot = rows * lda0;
      sa_build_kernel<<<(unsigned)((tot + 255) / 256), 256, 0, stream>>>(
          X[l], Cx, P[l], P[l + 1], nbr, m, lda0, ping);

      const _Float16* curA = ping;
      int cur_ld = lda0;
      bool in_ping = true;
      int Kd = Ct;
      for (int j = 0; j < 4; ++j) {
        const float* bp = (const float*)d_in[1 + l * 8 + 2 * j + 1];
        const int Nc = SA_C[l][j];
        const int ldc = pad32(Nc);
        _Float16* Co = in_ping ? pong : ping;
        launch_gemm<_Float16>(curA, cur_ld, wpSA[l][j], bp, Co, ldc,
                              (int)rows, Kd, Nc, 1, stream);
        curA = Co; in_ping = !in_ping; Kd = Nc; cur_ld = ldc;
      }
      maxpool_kernel<<<(unsigned)(((size_t)m * Kd + 255) / 256), 256, 0, stream>>>(
          curA, d2, R2[l], X[l + 1], m, Kd, 64);
    }

    // ---------------- FP levels (fp4 -> fp1)
    const float* inx = X[4]; int Cin = 512;
    const float* inp = P[4]; int nin = MS[4];
    float* fout = FA;
    const float* feats = FA;
    for (int s = 0; s < 4; ++s) {
      const int sl = 3 - s;
      const int ns = MS[sl], Cs = XCH[sl], k = FP_K[s];
      if (k == 1)
        knn_kernel<1><<<(ns + 127) / 128, 128, 0, stream>>>(P[sl], ns, inp, nin, nbr, d2);
      else
        knn_kernel<3><<<(ns + 127) / 128, 128, 0, stream>>>(P[sl], ns, inp, nin, nbr, d2);

      const int Ct = Cin + Cs + 3;
      const int lda0 = pad32(Ct);
      const size_t tot = (size_t)ns * lda0;
      fp_build_kernel<<<(unsigned)((tot + 255) / 256), 256, 0, stream>>>(
          inx, Cin, X[sl], Cs, P[sl], nbr, d2, k, ns, lda0, ping);

      const _Float16* curA = ping;
      int cur_ld = lda0;
      bool in_ping = true;
      int Kd = Ct;
      const int nl = FP_NL[s];
      for (int j = 0; j < nl; ++j) {
        const float* bp = (const float*)d_in[FP_BASE[s] + 2 * j + 1];
        const int Nc = FP_CHS[s][j];
        if (j == nl - 1) {
          launch_gemm<float>(curA, cur_ld, wpFP[s][j], bp, fout, Nc,
                             ns, Kd, Nc, 1, stream);
        } else {
          const int ldc = pad32(Nc);
          _Float16* Co = in_ping ? pong : ping;
          launch_gemm<_Float16>(curA, cur_ld, wpFP[s][j], bp, Co, ldc,
                                ns, Kd, Nc, 1, stream);
          curA = Co; in_ping = !in_ping; cur_ld = ldc;
        }
        Kd = Nc;
      }
      inx = fout; Cin = 256; inp = P[sl]; nin = ns;
      feats = fout;
      fout = (fout == FA) ? FB : FA;
    }

    // ---------------- classifiers (relu chain, bias only on last layer)
    cvt_f16_kernel<<<(NPTS * 256 + 255) / 256, 256, 0, stream>>>(feats, FEATH, NPTS * 256);
    for (int c = 0; c < 2; ++c) {
      launch_gemm<_Float16>(FEATH, 256, wpCLS[c][0], nullptr, CA, 128,
                            NPTS, 256, 128, 1, stream);
      launch_gemm<_Float16>(CA, 128, wpCLS[c][1], nullptr, CB, 64,
                            NPTS, 128, 64, 1, stream);
      launch_gemm<_Float16>(CB, 64, wpCLS[c][2], nullptr, CA, 32,
                            NPTS, 64, 32, 1, stream);
      launch_gemm<float>(CA, 32, wpCLS[c][3], (const float*)d_in[61 + c * 5 + 4],
                         LOG, 50, NPTS, 32, 50, 0, stream);
      mean50_kernel<<<(NPTS + 255) / 256, 256, 0, stream>>>(
          LOG, out + (size_t)c * B * NPTS + (size_t)b * NPTS, NPTS);
    }
  }
}
